// Attention_31404800868429
// MI455X (gfx1250) — matrix operand order repaired and checked by reading, not compiled, not run on hardware
//
#include <hip/hip_runtime.h>
#include <stdint.h>

// Problem constants (match reference)
#define B_    64
#define S_    512
#define DIM_  1024
#define COV_  256
#define NROWS (B_ * S_)   // 32768
#define G3    (3 * COV_)  // 768

typedef __bf16 bf16_t;
typedef __attribute__((ext_vector_type(16))) __bf16 v16bf;
typedef __attribute__((ext_vector_type(8)))  float  v8f;

union AFrag { unsigned int u[8]; v16bf v; };

// ---------- Tensor Data Mover (gfx1250) ----------
#if __has_builtin(__builtin_amdgcn_tensor_load_to_lds)
#define HAVE_TDM 1
typedef unsigned int v4u  __attribute__((ext_vector_type(4)));
typedef int          v8i_ __attribute__((ext_vector_type(8)));
typedef int          v4i_ __attribute__((ext_vector_type(4)));

// 2D tile load: tileDw0 dwords per row, tileRows rows, row stride strideDw0 dwords.
// D# layout per CDNA5 ISA (data_size=4B).
// 6-arg toolchain form: (g0, g1, g2, g3, g4, cpol) with groups 2/3 zero for 2D.
__device__ inline void tdm_load_2d(unsigned lds_addr, const void* gptr,
                                   unsigned tileDw0, unsigned tileRows,
                                   unsigned tensorDw0, unsigned strideDw0,
                                   unsigned tensorRows) {
  unsigned long long ga = (unsigned long long)(uintptr_t)gptr;
  v4u g0;
  g0[0] = 1u;                                                // count=1, user mode
  g0[1] = lds_addr;                                          // lds_addr[31:0]
  g0[2] = (unsigned)(ga & 0xFFFFFFFFu);                      // global_addr[31:0]
  g0[3] = (unsigned)((ga >> 32) & 0x01FFFFFFu) | (2u << 30); // global_addr[56:32] | type=2
  v8i_ g1;
  g1[0] = (int)(2u << 16);                                   // data_size=2 (4 bytes)
  g1[1] = (int)((tensorDw0 & 0xFFFFu) << 16);                // tensor_dim0[15:0] @ bit48
  g1[2] = (int)((tensorDw0 >> 16) | ((tensorRows & 0xFFFFu) << 16)); // dim0 hi | dim1 lo
  g1[3] = (int)((tensorRows >> 16) | (tileDw0 << 16));       // dim1 hi | tile_dim0
  g1[4] = (int)(tileRows & 0xFFFFu);                         // tile_dim1 (tile_dim2=0)
  g1[5] = (int)strideDw0;                                    // tensor_dim0_stride[31:0]
  g1[6] = 0;                                                 // stride0 hi | stride1 lo
  g1[7] = 0;
  v4i_ z4 = {0, 0, 0, 0};
  v8i_ z8 = {0, 0, 0, 0, 0, 0, 0, 0};
  __builtin_amdgcn_tensor_load_to_lds(g0, g1, z4, z4, z8, 0);
}
#else
#define HAVE_TDM 0
#endif

// ---------- bf16 helpers (round-to-nearest-even) ----------
__device__ inline unsigned int f2bf1(float f) {
  unsigned int u = __float_as_uint(f);
  return (u + 0x7FFFu + ((u >> 16) & 1u)) >> 16;
}
__device__ inline unsigned int f2bf2(float lo, float hi) {
  return f2bf1(lo) | (f2bf1(hi) << 16);
}

// ---------- WMMA fragment loaders ----------
// A: 16x32 bf16 row-major, ldDw = row stride in dwords.
// Lane L<16: M=L, K={0..7,16..23}; Lane L>=16: M=L-16, K={8..15,24..31}
__device__ inline v16bf a_frag(const unsigned int* base, int ldDw, int kbaseElems, int lane) {
  int m    = lane & 15;
  int koff = (kbaseElems >> 1) + ((lane & 16) ? 4 : 0);
  const unsigned int* row = base + m * ldDw + koff;
  AFrag f;
#pragma unroll
  for (int i = 0; i < 4; ++i) f.u[i] = row[i];
#pragma unroll
  for (int i = 0; i < 4; ++i) f.u[4 + i] = row[8 + i];
  return f.v;
}

// B: 32x16 bf16 pre-packed fragment-major: tile = 32 lanes * 8 dwords.
__device__ inline v16bf b_frag(const unsigned int* packed, int tile, int lane) {
  const unsigned int* p = packed + tile * 256 + lane * 8;
  AFrag f;
#pragma unroll
  for (int i = 0; i < 8; ++i) f.u[i] = p[i];
  return f.v;
}

// ---------- pack [N][srcStride] fp32 weight (first Kp cols) into B-frag tiles ----------
__global__ void pack_b_kernel(const float* __restrict__ W, int N, int Kp, int srcStride,
                              unsigned int* __restrict__ out) {
  int idx   = blockIdx.x * blockDim.x + threadIdx.x;
  int total = (Kp / 32) * (N / 16) * 256;
  if (idx >= total) return;
  int t      = idx >> 8;
  int lane   = (idx >> 3) & 31;
  int i      = idx & 7;
  int ntiles = N / 16;
  int ktile  = t / ntiles, ntile = t % ntiles;
  int n  = ntile * 16 + (lane & 15);
  int k0 = ktile * 32 + ((lane & 16) ? 8 : 0) + ((i & 4) ? 16 : 0) + 2 * (i & 3);
  out[idx] = f2bf2(W[n * srcStride + k0], W[n * srcStride + k0 + 1]);
}

// ---------- fp32 -> packed bf16 pairs ----------
__global__ void f32_to_bf16_kernel(const float* __restrict__ src,
                                   unsigned int* __restrict__ dst, int nDwords) {
  int i = blockIdx.x * blockDim.x + threadIdx.x;
  if (i >= nDwords) return;
  dst[i] = f2bf2(src[2 * i], src[2 * i + 1]);
}

// ---------- target = h @ W_in^T (64x1024, K=1024) ----------
__global__ void target_kernel(const float* __restrict__ h, const float* __restrict__ W_in,
                              float* __restrict__ target) {
  int idx = blockIdx.x * blockDim.x + threadIdx.x;
  int b = idx >> 10, n = idx & 1023;
  float acc = 0.f;
  for (int k = 0; k < DIM_; ++k) acc += h[b * DIM_ + k] * W_in[n * DIM_ + k];
  target[idx] = acc;
}

// ---------- ctx = context + cov @ W_cov^T + b_cov (WMMA bf16) ----------
// grid (2048, 8), block 256 (8 waves; one 16x16 N-tile per wave)
__global__ void gemm_ctx_kernel(const unsigned int* __restrict__ covB,
                                const unsigned int* __restrict__ pWcov,
                                const float* __restrict__ context,
                                const float* __restrict__ b_cov,
                                float* __restrict__ ctx_f32,
                                unsigned short* __restrict__ ctx_b16) {
  __shared__ unsigned int As[16 * 128]; // 16 rows x 256 bf16
  int tid  = threadIdx.x;
  int row0 = blockIdx.x * 16;

#if HAVE_TDM
  if (tid < 32) {
    tdm_load_2d((unsigned)(uintptr_t)As, covB + (size_t)row0 * 128,
                128, 16, 128, 128, NROWS);
    __builtin_amdgcn_s_wait_tensorcnt(0);
  }
#else
#pragma unroll
  for (int q = 0; q < 8; ++q) {
    int e = tid * 8 + q;
    As[e] = covB[(size_t)row0 * 128 + e];
  }
#endif
  __syncthreads();

  int wave = tid >> 5, lane = tid & 31;
  int ntile = blockIdx.y * 8 + wave;
  int n     = ntile * 16 + (lane & 15);
  int mbase = (lane & 16) ? 8 : 0;

  // prefetch epilogue operands while WMMAs run
#pragma unroll
  for (int i = 0; i < 8; ++i)
    __builtin_prefetch(&context[(size_t)(row0 + mbase + i) * DIM_ + n], 0, 0);

  v8f acc = {};
#pragma unroll
  for (int kt = 0; kt < 8; ++kt) {
    v16bf a = a_frag(As, 128, kt * 32, lane);
    v16bf b = b_frag(pWcov, kt * 64 + ntile, lane);
    acc = __builtin_amdgcn_wmma_f32_16x16x32_bf16(false, a, false, b, (short)0, acc, false, false);
  }
  float bc = b_cov[n];
#pragma unroll
  for (int i = 0; i < 8; ++i) {
    size_t r = (size_t)(row0 + mbase + i);
    float v  = acc[i] + context[r * DIM_ + n] + bc;
    ctx_f32[r * DIM_ + n] = v;
    ctx_b16[r * DIM_ + n] = (unsigned short)f2bf1(v);
  }
}

// ---------- attention scores (one wave per row) ----------
__global__ void attn_scores_kernel(const float* __restrict__ ctx, const float* __restrict__ target,
                                   const unsigned char* __restrict__ mask,
                                   float* __restrict__ attn_raw) {
  int wave = (blockIdx.x * blockDim.x + threadIdx.x) >> 5;
  int lane = threadIdx.x & 31;
  if (wave >= NROWS) return;
  int b = wave >> 9;
  float acc = 0.f;
  const float* crow = ctx + (size_t)wave * DIM_;
  const float* trow = target + b * DIM_;
  for (int d = lane; d < DIM_; d += 32) acc += crow[d] * trow[d];
#pragma unroll
  for (int off = 16; off > 0; off >>= 1) acc += __shfl_xor(acc, off, 32);
  if (lane == 0) attn_raw[wave] = mask[wave] ? -__builtin_huge_valf() : acc;
}

// ---------- softmax over S per batch ----------
__global__ void softmax_kernel(const float* __restrict__ attn_raw,
                               float* __restrict__ attn_ws, float* __restrict__ attn_out) {
  __shared__ float sm[512];
  int b = blockIdx.x, s = threadIdx.x;
  float x = attn_raw[b * S_ + s];
  sm[s] = x;
  __syncthreads();
  for (int off = 256; off > 0; off >>= 1) {
    if (s < off) sm[s] = fmaxf(sm[s], sm[s + off]);
    __syncthreads();
  }
  float mx = sm[0];
  __syncthreads();
  float e = __expf(x - mx);
  sm[s] = e;
  __syncthreads();
  for (int off = 256; off > 0; off >>= 1) {
    if (s < off) sm[s] += sm[s + off];
    __syncthreads();
  }
  float v = e / sm[0];
  attn_ws[b * S_ + s]  = v;
  attn_out[b * S_ + s] = v;
}

// ---------- wc[b,d] = sum_s attn[b,s]*ctx[b,s,d] ----------
__global__ void wc_kernel(const float* __restrict__ attn, const float* __restrict__ ctx,
                          float* __restrict__ wc) {
  int b = blockIdx.x;
  int d = blockIdx.y * blockDim.x + threadIdx.x;
  float acc = 0.f;
  const float* base = ctx + (size_t)b * S_ * DIM_ + d;
  const float* arow = attn + b * S_;
  for (int s = 0; s < S_; ++s) acc += arow[s] * base[(size_t)s * DIM_];
  wc[b * DIM_ + d] = acc;
}

// ---------- h_tilde = tanh(concat(wc,h) @ W_out^T) ----------
__global__ void h_tilde_kernel(const float* __restrict__ wc, const float* __restrict__ h,
                               const float* __restrict__ W_out, float* __restrict__ out) {
  int idx = blockIdx.x * blockDim.x + threadIdx.x;
  int b = idx >> 10, n = idx & 1023;
  float acc = 0.f;
  const float* wrow = W_out + (size_t)n * 2 * DIM_;
  for (int k = 0; k < DIM_; ++k) acc += wc[b * DIM_ + k] * wrow[k];
  for (int k = 0; k < DIM_; ++k) acc += h[b * DIM_ + k] * wrow[DIM_ + k];
  out[idx] = tanhf(acc);
}

// ---------- hpart[b,j] = b_ih[j] + h[b,:] . w_ih[j, 1024:2048] ----------
__global__ void hpart_kernel(const float* __restrict__ h, const float* __restrict__ w_ih,
                             const float* __restrict__ b_ih, float* __restrict__ hpart) {
  int idx = blockIdx.x * blockDim.x + threadIdx.x;
  if (idx >= B_ * G3) return;
  int b = idx / G3, j = idx % G3;
  float acc = b_ih[j];
  const float* wrow = w_ih + (size_t)j * (2 * DIM_ + 1) + DIM_;
  for (int k = 0; k < DIM_; ++k) acc += h[b * DIM_ + k] * wrow[k];
  hpart[idx] = acc;
}

// ---------- Fused GRU (WMMA bf16 + gate epilogue in LDS, TDM-staged A) ----------
// grid 2048 (16-row tile), block 256 (8 waves x 6 N-tiles = 768 cols)
__global__ void gru_fused_kernel(const unsigned int* __restrict__ ctxB,  // bf16 pairs [32768][512]
                                 const unsigned int* __restrict__ covB,  // bf16 pairs [32768][128]
                                 const unsigned int* __restrict__ pWih,  // packed B, K=1024,N=768
                                 const unsigned int* __restrict__ pWhh,  // packed B, K=256, N=768
                                 const float* __restrict__ w_ih,
                                 const float* __restrict__ b_hh,
                                 const float* __restrict__ hpart,
                                 const float* __restrict__ attn,
                                 const float* __restrict__ cov0,
                                 float* __restrict__ new_cov) {
  extern __shared__ char smem[];
  unsigned int* Actx = (unsigned int*)smem;        // 8192 dw (16 x 1024 bf16)
  unsigned int* Acov = Actx + 16 * 512;            // 2048 dw (16 x 256 bf16)
  float* Sum = (float*)(Acov + 16 * 128);          // 16*512
  float* Xn  = Sum + 16 * 512;                     // 16*256
  float* Hn  = Xn + 16 * 256;                      // 16*256

  int tid  = threadIdx.x;
  int row0 = blockIdx.x * 16;
  int bIdx = row0 >> 9; // all 16 rows share batch index (512 % 16 == 0)

#if HAVE_TDM
  if (tid < 32) {
    tdm_load_2d((unsigned)(uintptr_t)Actx, ctxB + (size_t)row0 * 512,
                512, 16, 512, 512, NROWS);
    tdm_load_2d((unsigned)(uintptr_t)Acov, covB + (size_t)row0 * 128,
                128, 16, 128, 128, NROWS);
    __builtin_amdgcn_s_wait_tensorcnt(0);
  }
#else
#pragma unroll
  for (int q = 0; q < 32; ++q) Actx[tid + 256 * q] = ctxB[(size_t)row0 * 512 + tid + 256 * q];
#pragma unroll
  for (int q = 0; q < 8; ++q)  Acov[tid + 256 * q] = covB[(size_t)row0 * 128 + tid + 256 * q];
#endif

  // prefetch gate-epilogue operand (cov0 tile: 16x256 fp32) while WMMAs run
  for (int e = tid; e < 16 * 256; e += 256)
    __builtin_prefetch(&cov0[(size_t)(row0 + (e >> 8)) * COV_ + (e & 255)], 0, 0);

  __syncthreads();

  int wave = tid >> 5, lane = tid & 31;
  int col  = lane & 15;
  int mb   = (lane & 16) ? 8 : 0;

  for (int q = 0; q < 6; ++q) {
    int nt = wave * 6 + q;        // 0..47
    int j  = nt * 16 + col;
    // gx over ctx (K = 1024)
    v8f ax = {};
#pragma unroll
    for (int kt = 0; kt < 32; ++kt) {
      v16bf a = a_frag(Actx, 512, kt * 32, lane);
      v16bf b = b_frag(pWih, kt * 48 + nt, lane);
      ax = __builtin_amdgcn_wmma_f32_16x16x32_bf16(false, a, false, b, (short)0, ax, false, false);
    }
    // gh over cov (K = 256)
    v8f ah = {};
#pragma unroll
    for (int kt = 0; kt < 8; ++kt) {
      v16bf a = a_frag(Acov, 128, kt * 32, lane);
      v16bf b = b_frag(pWhh, kt * 48 + nt, lane);
      ah = __builtin_amdgcn_wmma_f32_16x16x32_bf16(false, a, false, b, (short)0, ah, false, false);
    }
    float wlast = w_ih[(size_t)j * (2 * DIM_ + 1) + 2 * DIM_];
    float hp    = hpart[bIdx * G3 + j];
    float bhh   = b_hh[j];
#pragma unroll
    for (int i = 0; i < 8; ++i) {
      int m = mb + i;
      int r = row0 + m;
      float gx = ax[i] + hp + attn[r] * wlast;
      float gh = ah[i] + bhh;
      if (j < 512) {
        Sum[m * 512 + j] = gx + gh;            // r,z pre-activations
      } else {
        Xn[m * 256 + (j - 512)] = gx;
        Hn[m * 256 + (j - 512)] = gh;
      }
    }
  }
  __syncthreads();

  // gate epilogue: 16 rows x 256 cols
  for (int e = tid; e < 16 * 256; e += 256) {
    int m = e >> 8, j = e & 255;
    size_t r = (size_t)(row0 + m);
    float rg = 1.f / (1.f + __expf(-Sum[m * 512 + j]));
    float zg = 1.f / (1.f + __expf(-Sum[m * 512 + 256 + j]));
    float ng = tanhf(Xn[m * 256 + j] + rg * Hn[m * 256 + j]);
    float h0 = cov0[r * COV_ + j];
    new_cov[r * COV_ + j] = (1.f - zg) * ng + zg * h0;
  }
}

// ---------------------------------------------------------------------------
extern "C" void kernel_launch(void* const* d_in, const int* in_sizes, int n_in,
                              void* d_out, int out_size, void* d_ws, size_t ws_size,
                              hipStream_t stream) {
  const float* h       = (const float*)d_in[0];
  const float* context = (const float*)d_in[1];
  const float* cov     = (const float*)d_in[2];
  const float* W_in    = (const float*)d_in[3];
  const float* W_out   = (const float*)d_in[4];
  const float* W_cov   = (const float*)d_in[5];
  const float* b_cov   = (const float*)d_in[6];
  const float* w_ih    = (const float*)d_in[7];
  const float* w_hh    = (const float*)d_in[8];
  const float* b_ih    = (const float*)d_in[9];
  const float* b_hh    = (const float*)d_in[10];
  const unsigned char* mask = (const unsigned char*)d_in[11];

  float* out_htilde = (float*)d_out;               // 64*1024
  float* out_attn   = out_htilde + B_ * DIM_;      // 64*512
  float* out_newcov = out_attn + B_ * S_;          // 64*512*256

  char* ws = (char*)d_ws;
  size_t off = 0;
  auto carve = [&](size_t bytes) { void* p = ws + off; off += (bytes + 255) & ~(size_t)255; return p; };
  float*          ctx_f32  = (float*)carve((size_t)NROWS * DIM_ * 4);
  unsigned short* ctx_b16  = (unsigned short*)carve((size_t)NROWS * DIM_ * 2);
  unsigned int*   cov_b16  = (unsigned int*)carve((size_t)NROWS * COV_ * 2);
  unsigned int*   pWcov    = (unsigned int*)carve((size_t)COV_ * DIM_ * 2);
  unsigned int*   pWih     = (unsigned int*)carve((size_t)DIM_ * G3 * 2);
  unsigned int*   pWhh     = (unsigned int*)carve((size_t)COV_ * G3 * 2);
  float*          target   = (float*)carve((size_t)B_ * DIM_ * 4);
  float*          attn_raw = (float*)carve((size_t)NROWS * 4);
  float*          attn_ws  = (float*)carve((size_t)NROWS * 4);
  float*          wc       = (float*)carve((size_t)B_ * DIM_ * 4);
  float*          hpart    = (float*)carve((size_t)B_ * G3 * 4);

  // weight packing + input conversion
  pack_b_kernel<<<(8 * 64 * 256) / 256, 256, 0, stream>>>(W_cov, DIM_, COV_, COV_, pWcov);
  pack_b_kernel<<<(32 * 48 * 256) / 256, 256, 0, stream>>>(w_ih, G3, DIM_, 2 * DIM_ + 1, pWih);
  pack_b_kernel<<<(8 * 48 * 256) / 256, 256, 0, stream>>>(w_hh, G3, COV_, COV_, pWhh);
  f32_to_bf16_kernel<<<(NROWS * COV_ / 2) / 256, 256, 0, stream>>>(cov, cov_b16, NROWS * COV_ / 2);

  // small GEMMs
  target_kernel<<<256, 256, 0, stream>>>(h, W_in, target);
  hpart_kernel<<<(B_ * G3 + 255) / 256, 256, 0, stream>>>(h, w_ih, b_ih, hpart);

  // ctx GEMM (WMMA) -> ctx fp32 + bf16
  gemm_ctx_kernel<<<dim3(NROWS / 16, DIM_ / 128), 256, 0, stream>>>(
      cov_b16, pWcov, context, b_cov, ctx_f32, ctx_b16);

  // attention
  attn_scores_kernel<<<NROWS / 8, 256, 0, stream>>>(ctx_f32, target, mask, attn_raw);
  softmax_kernel<<<B_, S_, 0, stream>>>(attn_raw, attn_ws, out_attn);

  // weighted context + output projection
  wc_kernel<<<dim3(B_, DIM_ / 256), 256, 0, stream>>>(attn_ws, ctx_f32, wc);
  h_tilde_kernel<<<256, 256, 0, stream>>>(wc, h, W_out, out_htilde);

  // fused GRU (WMMA + gates)
  size_t gruLds = (16 * 512 + 16 * 128) * 4 + (16 * 512 + 16 * 256 + 16 * 256) * 4; // 106496 B
  (void)hipFuncSetAttribute((const void*)gru_fused_kernel,
                            hipFuncAttributeMaxDynamicSharedMemorySize, (int)gruLds);
  gru_fused_kernel<<<NROWS / 16, 256, gruLds, stream>>>(
      (const unsigned int*)ctx_b16, cov_b16, pWih, pWhh,
      w_ih, b_hh, hpart, attn_ws, cov, out_newcov);

  (void)in_sizes; (void)n_in; (void)out_size; (void)ws_size;
}
